// MultiHypothesisGraphTransformerLayer_3204045603787
// MI455X (gfx1250) — compile-verified, matrix operand order checked by reading, not run on hardware
//
#include <hip/hip_runtime.h>
#include <math.h>

// ---------------- problem constants ----------------
#define BATCH   2
#define EMBED   256
#define HEADS   8
#define HDIM    32
#define FFDIM   512
#define HH      24
#define WW      24
#define HWN     576           // H*W
#define PPD     6             // P = H/PATCH
#define P2      36            // P*P
#define NALL    1728          // 3*HW
#define XSZ     (BATCH*EMBED*HWN)      // 294912 floats per image tensor
#define SEQMAX  (NALL*BATCH*EMBED)     // 884736 floats: largest seq buffer
#define SSZ     ((size_t)2*NALL*NALL)  // per-head score buffer (both b)
#define ISQRT_HD 0.17677669529663687f  // 1/sqrt(32)

typedef __attribute__((ext_vector_type(16))) _Float16 v16h;
typedef __attribute__((ext_vector_type(8)))  _Float16 v8h;
typedef __attribute__((ext_vector_type(8)))  float    v8f;

// ---------------- wave32 reductions ----------------
__device__ __forceinline__ float waveSum(float v) {
#pragma unroll
  for (int o = 16; o > 0; o >>= 1) v += __shfl_xor(v, o, 32);
  return v;
}
__device__ __forceinline__ float waveMax(float v) {
#pragma unroll
  for (int o = 16; o > 0; o >>= 1) v = fmaxf(v, __shfl_xor(v, o, 32));
  return v;
}

// 8 contiguous-or-strided f32 loads into registers (no interleaved stores)
__device__ __forceinline__ void load8(const float* __restrict__ p, int stride, float* out) {
  if (stride == 1) {
    const float4 a = *(const float4*)p;
    const float4 b = *(const float4*)(p + 4);
    out[0] = a.x; out[1] = a.y; out[2] = a.z; out[3] = a.w;
    out[4] = b.x; out[5] = b.y; out[6] = b.z; out[7] = b.w;
  } else {
#pragma unroll
    for (int j = 0; j < 8; ++j) out[j] = p[(size_t)j * stride];
  }
}

// ---------------- generic strided batched WMMA GEMM ----------------
// C[z](m,n) = act( alpha * sum_k A[z](m,k)*B[z](n,k) + bias[n] + rb[rbIdx]*(E?E(m,n):1) )
// block = 128 threads (4 wave32), 32x32 output tile, 16x16 WMMA per wave, K step 32.
// f32 operands are converted to f16 while staging through LDS.
__launch_bounds__(128)
__global__ void gemm_wmma(const float* __restrict__ A, int sAm, int sAk, int sAz,
                          const float* __restrict__ B, int sBn, int sBk, int sBz,
                          float* __restrict__ C, int sCm, int sCn, int sCz,
                          const float* __restrict__ bias,
                          const float* __restrict__ Eg, int sEm, int sEn,
                          const float* __restrict__ rbP, int rbIdx,
                          float alpha, int relu,
                          int M, int N, int K)
{
  __shared__ _Float16 As[32][40];   // 40-halfword row stride: b128-aligned, low-conflict
  __shared__ _Float16 Bs[32][40];

  const int z  = blockIdx.z;
  const float* Ab = A + (size_t)z * sAz;
  const float* Bb = B + (size_t)z * sBz;
  float*       Cb = C + (size_t)z * sCz;

  const int tM = blockIdx.y << 5;
  const int tN = blockIdx.x << 5;
  const int tid  = threadIdx.x;
  const int lane = tid & 31;
  const int wave = tid >> 5;
  const int wm = (wave >> 1) << 4;   // wave's 16x16 sub-tile
  const int wn = (wave & 1) << 4;
  const int lr = lane & 15;          // row within 16 (M for A-frag, N for B-frag)
  const int lh = lane >> 4;          // K-half select (ISA 7.12.2 16-bit layout)

  // staging ownership: thread -> (row, 8-column group) of each 32x32 tile
  const int srow = tid >> 2;         // 0..31
  const int scol = (tid & 3) << 3;   // 0,8,16,24

  const float* apRow = Ab + (size_t)(tM + srow) * sAm + (size_t)scol * sAk;
  const float* bpRow = Bb + (size_t)(tN + srow) * sBn + (size_t)scol * sBk;
  const bool fullMN = (tM + 32 <= M) && (tN + 32 <= N);

  v8f acc = {};

  for (int kb = 0; kb < K; kb += 32) {
    v8h ah, bh;
    if (fullMN && (kb + 32 <= K)) {
      // ---- fast path: unguarded vector loads, then packed LDS stores ----
      float av[8], bv[8];
      load8(apRow, sAk, av);
      load8(bpRow, sBk, bv);
#pragma unroll
      for (int j = 0; j < 8; ++j) { ah[j] = (_Float16)av[j]; bh[j] = (_Float16)bv[j]; }
    } else {
      // ---- ragged tiles (M=36/72 global-GAT ops only) ----
      const int gm = tM + srow, gn = tN + srow;
#pragma unroll
      for (int j = 0; j < 8; ++j) {
        const int gk = kb + scol + j;
        ah[j] = (_Float16)((gm < M && gk < K) ? apRow[(size_t)j * sAk] : 0.0f);
        bh[j] = (_Float16)((gn < N && gk < K) ? bpRow[(size_t)j * sBk] : 0.0f);
      }
    }
    *(v8h*)(&As[srow][scol]) = ah;
    *(v8h*)(&Bs[srow][scol]) = bh;
    __syncthreads();

    v16h af, bf;
#pragma unroll
    for (int v = 0; v < 8; ++v) {
      // lanes 0-15: K = {0..7, 16..23}; lanes 16-31: K = {8..15, 24..31}
      const int k0 = ((v < 4) ? (v * 2) : (16 + (v - 4) * 2)) + lh * 8;
      af[2 * v]     = As[wm + lr][k0];
      af[2 * v + 1] = As[wm + lr][k0 + 1];
      bf[2 * v]     = Bs[wn + lr][k0];
      bf[2 * v + 1] = Bs[wn + lr][k0 + 1];
    }
    acc = __builtin_amdgcn_wmma_f32_16x16x32_f16(false, af, false, bf,
                                                 (short)0, acc, false, false);
    __syncthreads();

    apRow += (size_t)32 * sAk;
    bpRow += (size_t)32 * sBk;
  }

  // D layout: lane%16 -> N, VGPR r -> M = r + (lane/16)*8
#pragma unroll
  for (int r = 0; r < 8; ++r) {
    const int gm = tM + wm + lh * 8 + r;
    const int gn = tN + wn + lr;
    if (gm < M && gn < N) {
      float v = alpha * acc[r];
      if (bias) v += bias[gn];
      if (rbP)  v += rbP[rbIdx] * (Eg ? Eg[(size_t)gm * sEm + (size_t)gn * sEn] : 1.0f);
      if (relu) v = fmaxf(v, 0.0f);
      Cb[(size_t)gm * sCm + (size_t)gn * sCn] = v;
    }
  }
}

// ---------------- softmax over the batch axis (GAT quirk: softmax(dim=1 of hbij)) ----------------
__global__ void softmax_batch2(float* __restrict__ S, int MN) {
  const int idx = blockIdx.x * blockDim.x + threadIdx.x;
  if (idx < MN) {
    const float a = S[idx], b = S[MN + idx];
    const float m = fmaxf(a, b);
    const float ea = __expf(a - m), eb = __expf(b - m);
    const float inv = 1.0f / (ea + eb);
    S[idx] = ea * inv;
    S[MN + idx] = eb * inv;
  }
}

// ---------------- standard row softmax (MHA) ---------------- one wave per row
__global__ void softmax_row(float* __restrict__ S, int N) {
  float* p = S + (size_t)blockIdx.x * N;
  const int lane = threadIdx.x;
  float mx = -3.4e38f;
  for (int j = lane; j < N; j += 32) mx = fmaxf(mx, p[j]);
  mx = waveMax(mx);
  float s = 0.0f;
  for (int j = lane; j < N; j += 32) { const float e = __expf(p[j] - mx); p[j] = e; s += e; }
  s = waveSum(s);
  const float inv = 1.0f / s;
  for (int j = lane; j < N; j += 32) p[j] *= inv;
}

// ---------------- LayerNorm over last dim of token-major [T, EMBED]: Y = LN(A+R) ----------------
__global__ void ln_seq_add(const float* __restrict__ A, const float* __restrict__ R,
                           const float* __restrict__ ga, const float* __restrict__ be,
                           float* __restrict__ Y) {
  const int t = blockIdx.x, lane = threadIdx.x;
  const float* a = A + (size_t)t * EMBED;
  const float* r = R + (size_t)t * EMBED;
  float* y = Y + (size_t)t * EMBED;
  float v[8];
  float s = 0.0f;
#pragma unroll
  for (int i = 0; i < 8; ++i) { const int c = lane + i * 32; v[i] = a[c] + r[c]; s += v[i]; }
  s = waveSum(s);
  const float mean = s * (1.0f / EMBED);
  float var = 0.0f;
#pragma unroll
  for (int i = 0; i < 8; ++i) { const float d = v[i] - mean; var += d * d; }
  var = waveSum(var) * (1.0f / EMBED);
  const float inv = rsqrtf(var + 1e-5f);
#pragma unroll
  for (int i = 0; i < 8; ++i) {
    const int c = lane + i * 32;
    y[c] = (v[i] - mean) * inv * ga[c] + be[c];
  }
}

// ---------------- channel LayerNorm of [B,C,H,W]: Y = LN_c(X + unseq(Sq)) ----------------
__global__ void ln_chan_add(const float* __restrict__ X, const float* __restrict__ Sq,
                            const float* __restrict__ ga, const float* __restrict__ be,
                            float* __restrict__ Y, int nOff) {
  const int pos = blockIdx.x;
  const int b = pos / HWN, n = pos % HWN;
  const int lane = threadIdx.x;
  const float* sq = Sq + ((size_t)(n + nOff) * BATCH + b) * EMBED;
  float v[8];
  float s = 0.0f;
#pragma unroll
  for (int i = 0; i < 8; ++i) {
    const int c = lane + i * 32;
    v[i] = X[((size_t)b * EMBED + c) * HWN + n] + sq[c];
    s += v[i];
  }
  s = waveSum(s);
  const float mean = s * (1.0f / EMBED);
  float var = 0.0f;
#pragma unroll
  for (int i = 0; i < 8; ++i) { const float d = v[i] - mean; var += d * d; }
  var = waveSum(var) * (1.0f / EMBED);
  const float inv = rsqrtf(var + 1e-5f);
#pragma unroll
  for (int i = 0; i < 8; ++i) {
    const int c = lane + i * 32;
    Y[((size_t)b * EMBED + c) * HWN + n] = (v[i] - mean) * inv * ga[c] + be[c];
  }
}

// ---------------- [B,C,H,W] -> token-major seq [(n+nOff)*B + b][C] ----------------
__global__ void seq_copy(const float* __restrict__ X, float* __restrict__ Sq, int nOff) {
  const int idx = blockIdx.x * blockDim.x + threadIdx.x;
  if (idx < XSZ) {
    const int n = idx % HWN;
    const int c = (idx / HWN) % EMBED;
    const int b = idx / (HWN * EMBED);
    Sq[((size_t)(n + nOff) * BATCH + b) * EMBED + c] = X[idx];
  }
}

// ---------------- 4x4 average pool -> global super-node seq [pn*B+b][C] ----------------
__global__ void avgpool_seq(const float* __restrict__ X, float* __restrict__ G) {
  const int idx = blockIdx.x * blockDim.x + threadIdx.x;
  if (idx < P2 * BATCH * EMBED) {
    const int c = idx % EMBED;
    const int b = (idx / EMBED) % BATCH;
    const int pn = idx / (EMBED * BATCH);
    const int py = pn / PPD, px = pn % PPD;
    float s = 0.0f;
#pragma unroll
    for (int dy = 0; dy < 4; ++dy)
#pragma unroll
      for (int dx = 0; dx < 4; ++dx) {
        const int n = (py * 4 + dy) * WW + (px * 4 + dx);
        s += X[((size_t)b * EMBED + c) * HWN + n];
      }
    G[((size_t)pn * BATCH + b) * EMBED + c] = s * (1.0f / 16.0f);
  }
}

// ---------------- nearest-neighbor 4x upsample of global seq onto local grid ----------------
__global__ void upsample_seq(const float* __restrict__ G2, float* __restrict__ U) {
  const int idx = blockIdx.x * blockDim.x + threadIdx.x;
  if (idx < XSZ) {
    const int c = idx % EMBED;
    const int b = (idx / EMBED) % BATCH;
    const int n = idx / (EMBED * BATCH);
    const int y = n / WW, x = n % WW;
    const int gn = (y / 4) * PPD + (x / 4);
    U[idx] = G2[((size_t)gn * BATCH + b) * EMBED + c];
  }
}

// ---------------- 4-neighbor grid adjacency ----------------
__global__ void build_edges(float* __restrict__ E) {
  const int idx = blockIdx.x * blockDim.x + threadIdx.x;
  if (idx < HWN * HWN) {
    const int i = idx / HWN, j = idx % HWN;
    const int yi = i / WW, xi = i % WW, yj = j / WW, xj = j % WW;
    const int d = abs(yi - yj) + abs(xi - xj);
    E[idx] = (d == 1) ? 1.0f : 0.0f;
  }
}

// ==================== host side ====================
struct Lin { const float *b, *w; };
struct LnP { const float *b, *g; };
struct Gat { const float* rb; Lin wk, wo, wq, wv; };
struct Mha { const float *in_b, *in_w; Lin out; };
struct Hga { Gat g, l; LnP nc, ng, nl; Mha x; };
struct Cur { void* const* din; int i;
  const float* next() { return (const float*)din[i++]; } };

static Lin readLin(Cur& c) { Lin l; l.b = c.next(); l.w = c.next(); return l; }
static LnP readLn (Cur& c) { LnP l; l.b = c.next(); l.g = c.next(); return l; }
static Gat readGat(Cur& c) { Gat g; g.rb = c.next(); g.wk = readLin(c); g.wo = readLin(c);
                             g.wq = readLin(c); g.wv = readLin(c); return g; }
static Mha readMha(Cur& c) { Mha m; m.in_b = c.next(); m.in_w = c.next(); m.out = readLin(c); return m; }
static Hga readHga(Cur& c) { Hga h; h.g = readGat(c); h.l = readGat(c); h.nc = readLn(c);
                             h.ng = readLn(c); h.nl = readLn(c); h.x = readMha(c); return h; }

struct Bufs { float *Q, *K, *V, *S, *T1, *T2; };

static void gemmL(hipStream_t st,
                  const float* A, int sAm, int sAk, int sAz,
                  const float* B, int sBn, int sBk, int sBz,
                  float* C, int sCm, int sCn, int sCz,
                  const float* bias, const float* Eg, int sEm, int sEn,
                  const float* rbP, int rbIdx, float alpha, int relu,
                  int M, int N, int K, int nZ) {
  dim3 g((N + 31) / 32, (M + 31) / 32, nZ), blk(128);
  gemm_wmma<<<g, blk, 0, st>>>(A, sAm, sAk, sAz, B, sBn, sBk, sBz, C, sCm, sCn, sCz,
                               bias, Eg, sEm, sEn, rbP, rbIdx, alpha, relu, M, N, K);
}

// GAT on token-major nodes [Nn*B, EMBED]; writes projected output to b.T2.
static void gat_block(hipStream_t st, const Gat& g, const float* nodes, int Nn,
                      const float* Eg, int sE, Bufs& b) {
  const int T = Nn * BATCH;
  gemmL(st, nodes, EMBED, 1, 0, g.wq.w, EMBED, 1, 0, b.Q, EMBED, 1, 0,
        g.wq.b, nullptr, 0, 0, nullptr, 0, 1.0f, 0, T, EMBED, EMBED, 1);
  gemmL(st, nodes, EMBED, 1, 0, g.wk.w, EMBED, 1, 0, b.K, EMBED, 1, 0,
        g.wk.b, nullptr, 0, 0, nullptr, 0, 1.0f, 0, T, EMBED, EMBED, 1);
  gemmL(st, nodes, EMBED, 1, 0, g.wv.w, EMBED, 1, 0, b.V, EMBED, 1, 0,
        g.wv.b, nullptr, 0, 0, nullptr, 0, 1.0f, 0, T, EMBED, EMBED, 1);
  for (int h = 0; h < HEADS; ++h) {
    // scores S[b][i][j] = q.k/sqrt(hd) + rb[h]*E(i,j); z over b
    gemmL(st, b.Q + h * HDIM, BATCH * EMBED, 1, EMBED,
              b.K + h * HDIM, BATCH * EMBED, 1, EMBED,
              b.S, Nn, 1, Nn * Nn,
          nullptr, Eg, sE, 1, g.rb, h, ISQRT_HD, 0, Nn, Nn, HDIM, BATCH);
    softmax_batch2<<<(Nn * Nn + 255) / 256, 256, 0, st>>>(b.S, Nn * Nn);
    // o[i,b,h,:] = sum_j w[b][i][j] * v[j,b,h,:]
    gemmL(st, b.S, Nn, 1, Nn * Nn,
              b.V + h * HDIM, 1, BATCH * EMBED, EMBED,
              b.T1 + h * HDIM, BATCH * EMBED, 1, EMBED,
          nullptr, nullptr, 0, 0, nullptr, 0, 1.0f, 0, Nn, HDIM, Nn, BATCH);
  }
  gemmL(st, b.T1, EMBED, 1, 0, g.wo.w, EMBED, 1, 0, b.T2, EMBED, 1, 0,
        g.wo.b, nullptr, 0, 0, nullptr, 0, 1.0f, 0, T, EMBED, EMBED, 1);
}

// standard MHA (queries qn, keys/values kvn), output to b.T2
static void mha_block(hipStream_t st, const Mha& m, const float* qn, const float* kvn,
                      int Nn, Bufs& b) {
  const int T = Nn * BATCH;
  const float* wq = m.in_w;
  const float* wk = m.in_w + EMBED * EMBED;
  const float* wv = m.in_w + 2 * EMBED * EMBED;
  gemmL(st, qn, EMBED, 1, 0, wq, EMBED, 1, 0, b.Q, EMBED, 1, 0,
        m.in_b, nullptr, 0, 0, nullptr, 0, 1.0f, 0, T, EMBED, EMBED, 1);
  gemmL(st, kvn, EMBED, 1, 0, wk, EMBED, 1, 0, b.K, EMBED, 1, 0,
        m.in_b + EMBED, nullptr, 0, 0, nullptr, 0, 1.0f, 0, T, EMBED, EMBED, 1);
  gemmL(st, kvn, EMBED, 1, 0, wv, EMBED, 1, 0, b.V, EMBED, 1, 0,
        m.in_b + 2 * EMBED, nullptr, 0, 0, nullptr, 0, 1.0f, 0, T, EMBED, EMBED, 1);
  for (int h = 0; h < HEADS; ++h) {
    gemmL(st, b.Q + h * HDIM, BATCH * EMBED, 1, EMBED,
              b.K + h * HDIM, BATCH * EMBED, 1, EMBED,
              b.S, Nn, 1, Nn * Nn,
          nullptr, nullptr, 0, 0, nullptr, 0, ISQRT_HD, 0, Nn, Nn, HDIM, BATCH);
    softmax_row<<<BATCH * Nn, 32, 0, st>>>(b.S, Nn);
    gemmL(st, b.S, Nn, 1, Nn * Nn,
              b.V + h * HDIM, 1, BATCH * EMBED, EMBED,
              b.T1 + h * HDIM, BATCH * EMBED, 1, EMBED,
          nullptr, nullptr, 0, 0, nullptr, 0, 1.0f, 0, Nn, HDIM, Nn, BATCH);
  }
  gemmL(st, b.T1, EMBED, 1, 0, m.out.w, EMBED, 1, 0, b.T2, EMBED, 1, 0,
        m.out.b, nullptr, 0, 0, nullptr, 0, 1.0f, 0, T, EMBED, EMBED, 1);
}

static void hga_block(hipStream_t st, const float* xin, const Hga& hp, const LnP& n1,
                      float* xout, Bufs& b, float* L, float* L2, float* U,
                      float* G, float* G2, const float* EG) {
  avgpool_seq<<<(P2 * BATCH * EMBED + 255) / 256, 256, 0, st>>>(xin, G);
  gat_block(st, hp.g, G, P2, nullptr, 0, b);                      // all-ones edges via rb const
  ln_seq_add<<<P2 * BATCH, 32, 0, st>>>(b.T2, G, hp.ng.g, hp.ng.b, G2);
  seq_copy<<<(XSZ + 255) / 256, 256, 0, st>>>(xin, L, 0);
  gat_block(st, hp.l, L, HWN, EG, HWN, b);                        // grid edges
  ln_seq_add<<<HWN * BATCH, 32, 0, st>>>(b.T2, L, hp.nl.g, hp.nl.b, L2);
  upsample_seq<<<(XSZ + 255) / 256, 256, 0, st>>>(G2, U);
  mha_block(st, hp.x, L2, U, HWN, b);
  ln_seq_add<<<HWN * BATCH, 32, 0, st>>>(b.T2, L2, hp.nc.g, hp.nc.b, L); // co -> L
  ln_chan_add<<<BATCH * HWN, 32, 0, st>>>(xin, L, n1.g, n1.b, xout, 0);
}

extern "C" void kernel_launch(void* const* d_in, const int* in_sizes, int n_in,
                              void* d_out, int out_size, void* d_ws, size_t ws_size,
                              hipStream_t stream) {
  (void)in_sizes; (void)n_in; (void)out_size; (void)ws_size;
  const float* x[3] = {(const float*)d_in[0], (const float*)d_in[1], (const float*)d_in[2]};

  Cur c{d_in, 3};
  Lin ffn1 = readLin(c), ffn2 = readLin(c);
  Hga h[3]; h[0] = readHga(c); h[1] = readHga(c); h[2] = readHga(c);
  Gat inter = readGat(c);
  LnP n1 = readLn(c), n2 = readLn(c), n3 = readLn(c);

  // ---- workspace carve-up (floats) ----
  float* ws = (float*)d_ws;
  size_t off = 0;
  float* XB = ws + off; off += (size_t)3 * XSZ;           // updated x1..x3 (contiguous for final copy)
  float* NODES = ws + off; off += SEQMAX;
  Bufs b;
  b.Q  = ws + off; off += SEQMAX;
  b.K  = ws + off; off += SEQMAX;
  b.V  = ws + off; off += SEQMAX;
  b.T1 = ws + off; off += SEQMAX;
  b.T2 = ws + off; off += SEQMAX;
  b.S  = ws + off; off += SSZ;
  float* L  = ws + off; off += XSZ;
  float* L2 = ws + off; off += XSZ;
  float* U  = ws + off; off += XSZ;
  float* G  = ws + off; off += (size_t)P2 * BATCH * EMBED;
  float* G2 = ws + off; off += (size_t)P2 * BATCH * EMBED;
  float* EG = ws + off; off += (size_t)HWN * HWN;

  build_edges<<<(HWN * HWN + 255) / 256, 256, 0, stream>>>(EG);

  // ---- per-branch hierarchical graph attention + ln_chan(n1) ----
  for (int br = 0; br < 3; ++br)
    hga_block(stream, x[br], h[br], n1, XB + (size_t)br * XSZ, b, L, L2, U, G, G2, EG);

  // ---- inter-hypothesis GAT over 3*HW nodes (all-ones edges) + ln_chan(n2) ----
  for (int br = 0; br < 3; ++br)
    seq_copy<<<(XSZ + 255) / 256, 256, 0, stream>>>(XB + (size_t)br * XSZ, NODES, br * HWN);
  gat_block(stream, inter, NODES, NALL, nullptr, 0, b);
  for (int br = 0; br < 3; ++br)
    ln_chan_add<<<BATCH * HWN, 32, 0, stream>>>(XB + (size_t)br * XSZ, b.T2, n2.g, n2.b,
                                                XB + (size_t)br * XSZ, br * HWN);

  // ---- FFN (256 -> 512 relu -> 256) + ln_chan(n3) ----
  for (int br = 0; br < 3; ++br) {
    seq_copy<<<(XSZ + 255) / 256, 256, 0, stream>>>(XB + (size_t)br * XSZ, L, 0);
    gemmL(stream, L, EMBED, 1, 0, ffn1.w, EMBED, 1, 0, b.T1, FFDIM, 1, 0,
          ffn1.b, nullptr, 0, 0, nullptr, 0, 1.0f, 1, HWN * BATCH, FFDIM, EMBED, 1);
    gemmL(stream, b.T1, FFDIM, 1, 0, ffn2.w, FFDIM, 1, 0, b.T2, EMBED, 1, 0,
          ffn2.b, nullptr, 0, 0, nullptr, 0, 1.0f, 0, HWN * BATCH, EMBED, FFDIM, 1);
    ln_chan_add<<<BATCH * HWN, 32, 0, stream>>>(XB + (size_t)br * XSZ, b.T2, n3.g, n3.b,
                                                XB + (size_t)br * XSZ, 0);
  }

  // ---- concatenated outputs (x1,x2,x3 are contiguous in XB) ----
  hipMemcpyAsync(d_out, XB, (size_t)3 * XSZ * sizeof(float),
                 hipMemcpyDeviceToDevice, stream);
}
